// BMP_SNNodeBlock_38809324487022
// MI455X (gfx1250) — matrix operand order repaired and checked by reading, not compiled, not run on hardware
//
#include <hip/hip_runtime.h>
#include <hip/hip_fp16.h>
#include <cmath>

typedef __attribute__((ext_vector_type(16))) _Float16 v16h;
typedef __attribute__((ext_vector_type(8)))  float    v8f;

#define HDIM 128
#define BN_EPS 1e-5f

// Native CDNA5 float atomic max (no CAS loop). Non-returning form -> STOREcnt.
__device__ __forceinline__ void atomic_fmax_f32(float* p, float v) {
    asm volatile("global_atomic_max_num_f32 %0, %1, off" :: "v"(p), "v"(v) : "memory");
}

// ---------------------------------------------------------------------------
// K0: init fwd/bwd to -inf (segment-max identity), zero BN stat accumulators
// ---------------------------------------------------------------------------
__global__ void k_init(float* __restrict__ fwd, float* __restrict__ bwd,
                       float* __restrict__ stats, int nh) {
    int i = blockIdx.x * blockDim.x + threadIdx.x;
    if (i < nh) { fwd[i] = -INFINITY; bwd[i] = -INFINITY; }
    if (i < 4 * HDIM) stats[i] = 0.0f;
}

// ---------------------------------------------------------------------------
// K1: convert W (f32, row-major [Ktot x 128]) into f16 pre-packed in the exact
// WMMA B-fragment layout:
//   wp[ ((kb*128 + c)*2 + hi)*16 + h ] = W[kb*32 + hi*16 + h][c]
// so each lane's 16 halfs are one contiguous, 32B-aligned v16h load.
// ---------------------------------------------------------------------------
__global__ void k_pack_b(const float* __restrict__ w, _Float16* __restrict__ wp, int Ktot) {
    int i = blockIdx.x * blockDim.x + threadIdx.x;
    if (i >= Ktot * HDIM) return;
    int k = i / HDIM, c = i - k * HDIM;
    int kb = k >> 5, rem = k & 31;
    int hi = rem >> 4, h = rem & 15;
    wp[(((size_t)kb * HDIM + c) * 2 + hi) * 16 + h] = (_Float16)w[i];
}

// ---------------------------------------------------------------------------
// K2: scatter-max. One wave per edge: 32 lanes x float4 = 128 features.
//     8 native f32 atomic-max per lane (fwd[col], bwd[row]).
// ---------------------------------------------------------------------------
__global__ void k_scatter(const float* __restrict__ msg, const int* __restrict__ ei,
                          float* __restrict__ fwd, float* __restrict__ bwd, int E) {
    int wave = threadIdx.x >> 5;
    int lane = threadIdx.x & 31;
    int e = blockIdx.x * 8 + wave;
    if (e >= E) return;
    int r = ei[e];       // edge_index[0][e]
    int c = ei[E + e];   // edge_index[1][e]
    const float4* m4 = (const float4*)(msg + (size_t)e * HDIM);
    float4 v = m4[lane];
    float* fp = fwd + (size_t)c * HDIM + lane * 4;
    float* bp = bwd + (size_t)r * HDIM + lane * 4;
    atomic_fmax_f32(fp + 0, v.x); atomic_fmax_f32(fp + 1, v.y);
    atomic_fmax_f32(fp + 2, v.z); atomic_fmax_f32(fp + 3, v.w);
    atomic_fmax_f32(bp + 0, v.x); atomic_fmax_f32(bp + 1, v.y);
    atomic_fmax_f32(bp + 2, v.z); atomic_fmax_f32(bp + 3, v.w);
}

// ---------------------------------------------------------------------------
// K3/K5: WMMA GEMM  Y[n x 128] = A[n x Ktot] @ W[Ktot x 128] + bias,
// with fused per-column sum/sumsq accumulation for BatchNorm.
// A is the concat of up to 3 row-major f32 sources (each 128 wide).
// Block = 256 threads (8 waves): 16 rows x 128 cols; wave w owns cols [16w,16w+16).
// Inner loop: 2x ds_load_b128 (A frag) + 2x global_load_b128 (packed B) + wmma.
// ---------------------------------------------------------------------------
__global__ void k_gemm_bnstat(const float* __restrict__ A0, const float* __restrict__ A1,
                              const float* __restrict__ A2,
                              int w0, int w01, int Ktot,
                              const _Float16* __restrict__ Wp, const float* __restrict__ bias,
                              float* __restrict__ Y,
                              float* __restrict__ sum, float* __restrict__ sumsq, int n) {
    __shared__ __align__(16) _Float16 As[16 * (384 + 8)];   // padded rows, Ktot<=384
    const int ldk = Ktot + 8;
    const int m0  = blockIdx.x * 16;

    // Stage 16 x Ktot A-tile into LDS as f16 (with -inf -> 0 fixup matching
    // the reference's empty-segment-max semantics).
    for (int t = threadIdx.x; t < 16 * Ktot; t += blockDim.x) {
        int rr = t / Ktot, cc = t - rr * Ktot;
        int m = m0 + rr;
        float v = 0.0f;
        if (m < n) {
            if (cc < w0)       v = A0[(size_t)m * w0 + cc];
            else if (cc < w01) v = A1[(size_t)m * HDIM + (cc - w0)];
            else               v = A2[(size_t)m * HDIM + (cc - w01)];
            if (v == -INFINITY) v = 0.0f;
        }
        As[rr * ldk + cc] = (_Float16)v;
    }
    __syncthreads();

    const int wave = threadIdx.x >> 5;
    const int lane = threadIdx.x & 31;
    const int l    = lane & 15;
    const int hi   = lane >> 4;           // lane-half 0/1
    const int col  = wave * 16 + l;       // output column (N index)

    // Packed B: one contiguous v16h per (kstep, col, hi)
    const v16h* __restrict__ Bp = (const v16h*)Wp;

    v8f acc = {};
    const int nk = Ktot >> 5;             // K-steps of 32
    for (int kb = 0; kb < nk; ++kb) {
        const int k = kb * 32;
        v16h a;
        // A fragment (contiguous K, matching packed B): lanes 0-15: K 0-15;
        // lanes 16-31: K 16-31 (row M = lane&15). 8-half groups are
        // 16B-aligned -> ds_load_b128.
        const _Float16* arow = &As[l * ldk + k];
        #pragma unroll
        for (int i = 0; i < 8; ++i) {
            a[i]     = arow[hi * 16 + i];
            a[8 + i] = arow[hi * 16 + 8 + i];
        }
        v16h b = Bp[((size_t)kb * HDIM + col) * 2 + hi];

        acc = __builtin_amdgcn_wmma_f32_16x16x32_f16(
            /*neg_a=*/false, a, /*neg_b=*/false, b,
            /*c_mod=*/(short)0, acc, /*reuse_a=*/false, /*reuse_b=*/false);
    }

    // Epilogue: +bias, store, accumulate BN stats (biased, valid rows only).
    const float bs = bias[col];
    float s = 0.0f, sq = 0.0f;
    #pragma unroll
    for (int r = 0; r < 8; ++r) {
        int m = m0 + hi * 8 + r;          // C layout: VGPR r -> row hi*8+r
        if (m < n) {
            float v = acc[r] + bs;
            Y[(size_t)m * HDIM + col] = v;
            s += v; sq += v * v;
        }
    }
    // lanes l and l+16 share a column: combine, then only low half-wave commits.
    s  += __shfl_xor(s, 16, 32);
    sq += __shfl_xor(sq, 16, 32);
    if (hi == 0) {
        atomicAdd(&sum[col], s);
        atomicAdd(&sumsq[col], sq);
    }
}

// ---------------------------------------------------------------------------
// K4: elementwise BN (training stats) + ReLU
// ---------------------------------------------------------------------------
__global__ void k_bnrelu(const float* __restrict__ Y, const float* __restrict__ sum,
                         const float* __restrict__ sumsq, const float* __restrict__ g,
                         const float* __restrict__ be, float* __restrict__ Out, int n) {
    int i = blockIdx.x * blockDim.x + threadIdx.x;
    if (i >= n * HDIM) return;
    int c = i & (HDIM - 1);
    float inv_n = 1.0f / (float)n;
    float mu  = sum[c] * inv_n;
    float var = sumsq[c] * inv_n - mu * mu;
    float v = (Y[i] - mu) * rsqrtf(var + BN_EPS) * g[c] + be[c];
    Out[i] = v > 0.0f ? v : 0.0f;
}

// ---------------------------------------------------------------------------
// K6: BN2 + ReLU in place in d_out, fused attention head:
//     att = sigmoid(h . Wa + ba). One wave per node; __shfl_xor reduction.
// ---------------------------------------------------------------------------
__global__ void k_bn2_att(float* __restrict__ hbuf, const float* __restrict__ sum,
                          const float* __restrict__ sumsq, const float* __restrict__ g,
                          const float* __restrict__ be, const float* __restrict__ Wa,
                          const float* __restrict__ ba, float* __restrict__ att, int n) {
    int wave = threadIdx.x >> 5;
    int lane = threadIdx.x & 31;
    int node = blockIdx.x * 8 + wave;
    if (node >= n) return;
    float inv_n = 1.0f / (float)n;
    float acc = 0.0f;
    #pragma unroll
    for (int j = 0; j < 4; ++j) {
        int c = lane + 32 * j;
        float mu  = sum[c] * inv_n;
        float var = sumsq[c] * inv_n - mu * mu;
        size_t idx = (size_t)node * HDIM + c;
        float v = (hbuf[idx] - mu) * rsqrtf(var + BN_EPS) * g[c] + be[c];
        v = v > 0.0f ? v : 0.0f;
        hbuf[idx] = v;
        acc += v * Wa[c];
    }
    #pragma unroll
    for (int m = 16; m >= 1; m >>= 1) acc += __shfl_xor(acc, m, 32);
    if (lane == 0) att[node] = 1.0f / (1.0f + expf(-(acc + ba[0])));
}

// ---------------------------------------------------------------------------
extern "C" void kernel_launch(void* const* d_in, const int* in_sizes, int n_in,
                              void* d_out, int out_size, void* d_ws, size_t ws_size,
                              hipStream_t stream) {
    const float* x   = (const float*)d_in[0];
    const float* msg = (const float*)d_in[1];
    const float* W1  = (const float*)d_in[2];
    const float* b1  = (const float*)d_in[3];
    const float* g1  = (const float*)d_in[4];
    const float* be1 = (const float*)d_in[5];
    const float* W2  = (const float*)d_in[6];
    const float* b2  = (const float*)d_in[7];
    const float* g2  = (const float*)d_in[8];
    const float* be2 = (const float*)d_in[9];
    const float* Wa  = (const float*)d_in[10];
    const float* ba  = (const float*)d_in[11];
    const int*   ei  = (const int*)d_in[12];

    const int N = in_sizes[0] / HDIM;
    const int E = in_sizes[12] / 2;
    const size_t nh = (size_t)N * HDIM;

    // Workspace layout (all offsets stay 32B-aligned)
    float*    fwd   = (float*)d_ws;          // N*H   (later reused as a2)
    float*    bwd   = fwd + nh;              // N*H
    float*    y1    = bwd + nh;              // N*H   (pre-BN GEMM1 output)
    float*    stats = y1 + nh;               // 4*H: sum1, sumsq1, sum2, sumsq2
    _Float16* W1p   = (_Float16*)(stats + 4 * HDIM);   // 384*128 f16 (packed)
    _Float16* W2p   = W1p + 384 * HDIM;                // 128*128 f16 (packed)

    float* sum1 = stats, *sumsq1 = stats + HDIM;
    float* sum2 = stats + 2 * HDIM, *sumsq2 = stats + 3 * HDIM;

    float* hout = (float*)d_out;             // N*H
    float* att  = hout + nh;                 // N

    const int nhInt = (int)nh;

    k_init<<<(nhInt + 255) / 256, 256, 0, stream>>>(fwd, bwd, stats, nhInt);
    k_pack_b<<<(384 * HDIM + 255) / 256, 256, 0, stream>>>(W1, W1p, 384);
    k_pack_b<<<(HDIM * HDIM + 255) / 256, 256, 0, stream>>>(W2, W2p, HDIM);

    k_scatter<<<(E + 7) / 8, 256, 0, stream>>>(msg, ei, fwd, bwd, E);

    // GEMM1: A = concat(x, fwd, bwd)  [N x 384]
    k_gemm_bnstat<<<(N + 15) / 16, 256, 0, stream>>>(
        x, fwd, bwd, HDIM, 2 * HDIM, 3 * HDIM, W1p, b1, y1, sum1, sumsq1, N);

    // BN1 + ReLU -> a2 (reuse fwd buffer)
    k_bnrelu<<<(nhInt + 255) / 256, 256, 0, stream>>>(y1, sum1, sumsq1, g1, be1, fwd, N);

    // GEMM2: A = a2 [N x 128], pre-BN result straight into d_out
    k_gemm_bnstat<<<(N + 15) / 16, 256, 0, stream>>>(
        fwd, nullptr, nullptr, HDIM, HDIM, HDIM, W2p, b2, hout, sum2, sumsq2, N);

    // BN2 + ReLU in place + attention
    k_bn2_att<<<(N + 7) / 8, 256, 0, stream>>>(hout, sum2, sumsq2, g2, be2, Wa, ba, att, N);
}
